// DLRM_Net_78116865179737
// MI455X (gfx1250) — compile-verified, matrix operand order checked by reading, not run on hardware
//
#include <hip/hip_runtime.h>
#include <hip/hip_bf16.h>
#include <math.h>

// ---------------------------------------------------------------------------
// DLRM forward on gfx1250: all GEMMs via v_wmma_f32_16x16x32_f16 (wave32).
// ---------------------------------------------------------------------------

typedef __attribute__((ext_vector_type(16))) _Float16 v16h;
typedef __attribute__((ext_vector_type(8)))  _Float16 v8h;
typedef __attribute__((ext_vector_type(4)))  _Float16 v4h;
typedef __attribute__((ext_vector_type(8)))  float    v8f;

static constexpr int BATCH   = 32768;
static constexpr int NTAB    = 26;
static constexpr int NROWS   = 100000;
static constexpr int DEMB    = 128;
static constexpr int TROWS   = 27;          // 1 dense + 26 embeddings
static constexpr int TLD     = TROWS * DEMB; // 3456 halves per sample
static constexpr int RLD     = 480;          // 128 + 351 + 1 pad

union V16H { v16h v; v8h h[2]; };

// 16-bit WMMA operand layout (A rows / B columns are symmetric):
//   lanes 0-15 : K = [k..k+7] , [k+16..k+23]
//   lanes 16-31: K = [k+8..k+15], [k+24..k+31]
__device__ __forceinline__ v16h load_tile_row(const _Float16* p, int lane) {
    const int ksel = (lane & 16) ? 8 : 0;
    V16H r;
    r.h[0] = *(const v8h*)(p + ksel);
    r.h[1] = *(const v8h*)(p + 16 + ksel);
    return r.v;
}

// ---------------------------------------------------------------------------
// f32 -> f16 conversion with K padding (used for weights and dense input)
// ---------------------------------------------------------------------------
__global__ void cvt_f32_to_f16_pad(const float* __restrict__ src,
                                   _Float16* __restrict__ dst,
                                   int M, int Kin, int Kpad) {
    long tid = (long)blockIdx.x * blockDim.x + threadIdx.x;
    if (tid >= (long)M * Kpad) return;
    int k = (int)(tid % Kpad);
    int m = (int)(tid / Kpad);
    dst[tid] = (k < Kin) ? (_Float16)src[(size_t)m * Kin + k] : (_Float16)0.f;
}

// ---------------------------------------------------------------------------
// Embedding gather: emb_tables (26,100000,128) f32 -> T[b][1+t][:] f16
// ---------------------------------------------------------------------------
__global__ void gather_embed(const float* __restrict__ emb,
                             const int* __restrict__ idx,
                             _Float16* __restrict__ Tmat) {
    long tid = (long)blockIdx.x * blockDim.x + threadIdx.x;
    if (tid >= (long)BATCH * NTAB * 32) return;
    int d4 = (int)(tid & 31);
    long bt = tid >> 5;
    int t = (int)(bt % NTAB);
    int b = (int)(bt / NTAB);
    int row = idx[(size_t)t * BATCH + b];
    const float* src = emb + ((size_t)t * NROWS + row) * DEMB + d4 * 4;
    float4 f = *(const float4*)src;
    v4h h;
    h.x = (_Float16)f.x; h.y = (_Float16)f.y;
    h.z = (_Float16)f.z; h.w = (_Float16)f.w;
    *(v4h*)(Tmat + (size_t)b * TLD + (t + 1) * DEMB + d4 * 4) = h;
}

// ---------------------------------------------------------------------------
// Y = relu(X(BxK f16, ld=K) @ W^T (W: MxK f16, ld=K) + bias) -> f16, ld=ldY
// One wave computes a 16x64 output tile (A reused across 4 B tiles).
// ---------------------------------------------------------------------------
__launch_bounds__(256)
__global__ void gemm_xwt_relu(const _Float16* __restrict__ X,
                              const _Float16* __restrict__ W,
                              const float* __restrict__ bias,
                              _Float16* __restrict__ Y,
                              int K, int ldY, int tilesM) {
    const int lane    = threadIdx.x & 31;
    const int wave    = blockIdx.x * (blockDim.x >> 5) + (threadIdx.x >> 5);
    const int rowTile = (wave / tilesM) * 16;
    const int colBase = (wave % tilesM) * 64;

    const _Float16* xr = X + (size_t)(rowTile + (lane & 15)) * K;
    const _Float16* w0 = W + (size_t)(colBase +  0 + (lane & 15)) * K;
    const _Float16* w1 = W + (size_t)(colBase + 16 + (lane & 15)) * K;
    const _Float16* w2 = W + (size_t)(colBase + 32 + (lane & 15)) * K;
    const _Float16* w3 = W + (size_t)(colBase + 48 + (lane & 15)) * K;

    v8f acc0 = {}, acc1 = {}, acc2 = {}, acc3 = {};
    for (int k = 0; k < K; k += 32) {
        v16h a  = load_tile_row(xr + k, lane);
        v16h b0 = load_tile_row(w0 + k, lane);
        v16h b1 = load_tile_row(w1 + k, lane);
        v16h b2 = load_tile_row(w2 + k, lane);
        v16h b3 = load_tile_row(w3 + k, lane);
        acc0 = __builtin_amdgcn_wmma_f32_16x16x32_f16(false, a, false, b0, (short)0, acc0, false, false);
        acc1 = __builtin_amdgcn_wmma_f32_16x16x32_f16(false, a, false, b1, (short)0, acc1, false, false);
        acc2 = __builtin_amdgcn_wmma_f32_16x16x32_f16(false, a, false, b2, (short)0, acc2, false, false);
        acc3 = __builtin_amdgcn_wmma_f32_16x16x32_f16(false, a, false, b3, (short)0, acc3, false, false);
    }

    // C/D layout: VGPR r -> row rowTile + r + (lane>=16 ? 8 : 0), col = lane&15
    const int rowOut = rowTile + ((lane & 16) ? 8 : 0);
    v8f accs[4] = {acc0, acc1, acc2, acc3};
#pragma unroll
    for (int t = 0; t < 4; ++t) {
        int col = colBase + t * 16 + (lane & 15);
        float bv = bias[col];
#pragma unroll
        for (int r = 0; r < 8; ++r) {
            float v = accs[t][r] + bv;
            v = v > 0.f ? v : 0.f;
            Y[(size_t)(rowOut + r) * ldY + col] = (_Float16)v;
        }
    }
}

// ---------------------------------------------------------------------------
// Interaction: per sample Z = T * T^T (27x27), store strict lower triangle
// into R[b][128..478]; also copy dense x into R[b][0..127], zero pad col 479.
// B-operand columns of T^T are T rows -> identical register image as A.
// ---------------------------------------------------------------------------
__launch_bounds__(256)
__global__ void interact(const _Float16* __restrict__ Tmat,
                         _Float16* __restrict__ R) {
    const int lane = threadIdx.x & 31;
    const int b    = blockIdx.x * (blockDim.x >> 5) + (threadIdx.x >> 5);
    const _Float16* T = Tmat + (size_t)b * TLD;
    _Float16* r = R + (size_t)b * RLD;

    *(v4h*)(r + lane * 4) = *(const v4h*)(T + lane * 4);  // copy x (128 halves)
    if (lane == 0) r[479] = (_Float16)0.f;                // K pad column

    const _Float16* t0 = T + (size_t)(lane & 15) * DEMB;
    const _Float16* t1 = T + (size_t)((lane & 15) + 16) * DEMB;
    v8f z00 = {}, z10 = {}, z11 = {};
    for (int k = 0; k < DEMB; k += 32) {
        v16h a0 = load_tile_row(t0 + k, lane);
        v16h a1 = load_tile_row(t1 + k, lane);
        z00 = __builtin_amdgcn_wmma_f32_16x16x32_f16(false, a0, false, a0, (short)0, z00, false, false);
        z10 = __builtin_amdgcn_wmma_f32_16x16x32_f16(false, a1, false, a0, (short)0, z10, false, false);
        z11 = __builtin_amdgcn_wmma_f32_16x16x32_f16(false, a1, false, a1, (short)0, z11, false, false);
    }

    const int jj    = lane & 15;
    const int ibase = (lane & 16) ? 8 : 0;
#pragma unroll
    for (int rr = 0; rr < 8; ++rr) {
        int i = ibase + rr;                       // tile (0,0)
        if (jj < i) r[128 + i * (i - 1) / 2 + jj] = (_Float16)z00[rr];
        i = 16 + ibase + rr;
        if (i < TROWS) {
            r[128 + i * (i - 1) / 2 + jj] = (_Float16)z10[rr];        // tile (1,0)
            int j2 = 16 + jj;                                          // tile (1,1)
            if (j2 < i) r[128 + i * (i - 1) / 2 + j2] = (_Float16)z11[rr];
        }
    }
}

// ---------------------------------------------------------------------------
// Final layer: out[b] = sigmoid(dot(H[b][0..255], W) + bias)  (wave32 reduce)
// ---------------------------------------------------------------------------
__launch_bounds__(256)
__global__ void final_layer(const _Float16* __restrict__ H,
                            const float* __restrict__ W,
                            const float* __restrict__ bias,
                            float* __restrict__ out) {
    const int lane = threadIdx.x & 31;
    const int b    = blockIdx.x * (blockDim.x >> 5) + (threadIdx.x >> 5);
    const _Float16* h = H + (size_t)b * 256 + lane * 8;
    const float* w = W + lane * 8;
    float s = 0.f;
#pragma unroll
    for (int i = 0; i < 8; ++i) s += (float)h[i] * w[i];
#pragma unroll
    for (int off = 16; off >= 1; off >>= 1) s += __shfl_xor(s, off, 32);
    if (lane == 0) out[b] = 1.0f / (1.0f + expf(-(s + bias[0])));
}

// ---------------------------------------------------------------------------
// Host launcher
// ---------------------------------------------------------------------------
extern "C" void kernel_launch(void* const* d_in, const int* in_sizes, int n_in,
                              void* d_out, int out_size, void* d_ws, size_t ws_size,
                              hipStream_t stream) {
    (void)in_sizes; (void)out_size; (void)ws_size;
    if (n_in < 19) return;

    // Inputs (dict order, tuples flattened)
    const float* dense_x = (const float*)d_in[0];
    const int*   lS_i    = (const int*)d_in[1];
    const float* emb     = (const float*)d_in[2];
    const float* bW[3]   = {(const float*)d_in[3], (const float*)d_in[4], (const float*)d_in[5]};
    const float* bB[3]   = {(const float*)d_in[6], (const float*)d_in[7], (const float*)d_in[8]};
    const float* tW[5]   = {(const float*)d_in[9], (const float*)d_in[10], (const float*)d_in[11],
                            (const float*)d_in[12], (const float*)d_in[13]};
    const float* tB[5]   = {(const float*)d_in[14], (const float*)d_in[15], (const float*)d_in[16],
                            (const float*)d_in[17], (const float*)d_in[18]};
    float* out = (float*)d_out;

    // Workspace layout
    char* ws = (char*)d_ws;
    size_t off = 0;
    auto alloc = [&](size_t bytes) -> char* {
        char* p = ws + off;
        off = (off + bytes + 255) & ~(size_t)255;
        return p;
    };
    _Float16* wb0 = (_Float16*)alloc((size_t)512 * 32 * 2);        // bot W0 padded 13->32
    _Float16* wb1 = (_Float16*)alloc((size_t)256 * 512 * 2);
    _Float16* wb2 = (_Float16*)alloc((size_t)128 * 256 * 2);
    _Float16* wt0 = (_Float16*)alloc((size_t)1024 * 480 * 2);      // top W0 padded 479->480
    _Float16* wt1 = (_Float16*)alloc((size_t)1024 * 1024 * 2);
    _Float16* wt2 = (_Float16*)alloc((size_t)512 * 1024 * 2);
    _Float16* wt3 = (_Float16*)alloc((size_t)256 * 512 * 2);
    _Float16* Xd  = (_Float16*)alloc((size_t)BATCH * 32 * 2);      // dense padded 13->32
    _Float16* A1  = (_Float16*)alloc((size_t)BATCH * 512 * 2);
    _Float16* A2  = (_Float16*)alloc((size_t)BATCH * 256 * 2);
    _Float16* Tm  = (_Float16*)alloc((size_t)BATCH * TLD * 2 + 4096); // +slack for row-32 overread
    _Float16* R   = (_Float16*)alloc((size_t)BATCH * RLD * 2);
    // T is dead once R exists; alias H1/H2 into its region. A1/A2 dead after bottom MLP.
    _Float16* H1 = Tm;
    _Float16* H2 = (_Float16*)((char*)Tm + (size_t)BATCH * 1024 * 2);
    _Float16* H3 = A1;
    _Float16* H4 = A2;

    const int TPB = 256;
    auto blocks = [](long n, int tpb) { return (unsigned)((n + tpb - 1) / tpb); };

    // 1) Convert weights + dense input to f16 (K-padded)
    cvt_f32_to_f16_pad<<<blocks((long)BATCH * 32, TPB), TPB, 0, stream>>>(dense_x, Xd, BATCH, 13, 32);
    cvt_f32_to_f16_pad<<<blocks((long)512 * 32, TPB),   TPB, 0, stream>>>(bW[0], wb0, 512, 13, 32);
    cvt_f32_to_f16_pad<<<blocks((long)256 * 512, TPB),  TPB, 0, stream>>>(bW[1], wb1, 256, 512, 512);
    cvt_f32_to_f16_pad<<<blocks((long)128 * 256, TPB),  TPB, 0, stream>>>(bW[2], wb2, 128, 256, 256);
    cvt_f32_to_f16_pad<<<blocks((long)1024 * 480, TPB), TPB, 0, stream>>>(tW[0], wt0, 1024, 479, 480);
    cvt_f32_to_f16_pad<<<blocks((long)1024 * 1024, TPB),TPB, 0, stream>>>(tW[1], wt1, 1024, 1024, 1024);
    cvt_f32_to_f16_pad<<<blocks((long)512 * 1024, TPB), TPB, 0, stream>>>(tW[2], wt2, 512, 1024, 1024);
    cvt_f32_to_f16_pad<<<blocks((long)256 * 512, TPB),  TPB, 0, stream>>>(tW[3], wt3, 256, 512, 512);

    // 2) Bottom MLP (last layer writes straight into T row 0, ldY = 27*128)
    const int rowWaves = BATCH / 16;  // 2048
    auto gemm_blocks = [&](int tilesM) { return (unsigned)(rowWaves * tilesM / 8); };
    gemm_xwt_relu<<<gemm_blocks(8),  TPB, 0, stream>>>(Xd, wb0, bB[0], A1, 32, 512, 8);
    gemm_xwt_relu<<<gemm_blocks(4),  TPB, 0, stream>>>(A1, wb1, bB[1], A2, 512, 256, 4);
    gemm_xwt_relu<<<gemm_blocks(2),  TPB, 0, stream>>>(A2, wb2, bB[2], Tm, 256, TLD, 2);

    // 3) Embedding gather into T rows 1..26
    gather_embed<<<blocks((long)BATCH * NTAB * 32, TPB), TPB, 0, stream>>>(emb, lS_i, Tm);

    // 4) Pairwise interaction -> R (B x 480, col 479 zero pad)
    interact<<<(unsigned)(BATCH / 8), TPB, 0, stream>>>(Tm, R);

    // 5) Top MLP
    gemm_xwt_relu<<<gemm_blocks(16), TPB, 0, stream>>>(R,  wt0, tB[0], H1, 480, 1024, 16);
    gemm_xwt_relu<<<gemm_blocks(16), TPB, 0, stream>>>(H1, wt1, tB[1], H2, 1024, 1024, 16);
    gemm_xwt_relu<<<gemm_blocks(8),  TPB, 0, stream>>>(H2, wt2, tB[2], H3, 1024, 512, 8);
    gemm_xwt_relu<<<gemm_blocks(4),  TPB, 0, stream>>>(H3, wt3, tB[3], H4, 512, 256, 4);

    // 6) Final 256->1 + sigmoid
    final_layer<<<(unsigned)(BATCH / 8), TPB, 0, stream>>>(H4, tW[4], tB[4], out);
}
